// BlockDecomposedSSMAttention_26173530701943
// MI455X (gfx1250) — matrix operand order repaired and checked by reading, not compiled, not run on hardware
//
#include <hip/hip_runtime.h>
#include <hip/hip_bf16.h>
#include <stdint.h>

// ---------------------------------------------------------------------------
// y = x @ (B^T A C^T):  collapse the 3-GEMM chain into W = B^T A C^T (1024^2),
// then one streaming GEMM y = x @ W.
// bf16 WMMA (v_wmma_f32_16x16x32_bf16), f32 accumulate.
// Tile staging uses CDNA5 async-to-LDS (GLOBAL_LOAD_ASYNC_TO_LDS_B128,
// ASYNCcnt) with double-buffered LDS; all B operands are pre-transposed
// (N x K) by the conversion passes so no transpose happens in the hot loop.
// ---------------------------------------------------------------------------

typedef __bf16 bf16_t;
typedef __attribute__((ext_vector_type(16))) __bf16 v16bf;
typedef __attribute__((ext_vector_type(8)))  float  v8f;

#define BM 128
#define BN 128
#define BK 32
#define PAD 8            // 16B LDS row pad: 16 fragment rows -> 16 distinct banks
#define LDK (BK + PAD)   // 40 bf16 = 80B row stride
#define THREADS 256      // 8 wave32s

union FragU { v16bf v; uint4 q[2]; };

// D(f32, MxN) = A(bf16, MxK row-major) * BT(bf16, NxK row-major)
// M, N multiples of 128; K multiple of 32.
__global__ __launch_bounds__(THREADS)
void gemm_bf16_wmma(const bf16_t* __restrict__ A,
                    const bf16_t* __restrict__ BT,
                    float* __restrict__ D,
                    int M, int N, int K)
{
    __shared__ bf16_t sA [2][BM][LDK];   // A tile,  [m][k]
    __shared__ bf16_t sBT[2][BN][LDK];   // B tile,  [n][k] (already transposed)

    const int tid  = threadIdx.x;
    const int lane = tid & 31;
    const int wave = tid >> 5;
    const int wm   = (wave & 3) * 32;    // wave M offset inside block tile
    const int wn   = (wave >> 2) * 64;   // wave N offset inside block tile

    const int bm = blockIdx.x * BM;
    const int bn = blockIdx.y * BN;

    v8f acc[2][4];
    const v8f vzero = {0.f,0.f,0.f,0.f,0.f,0.f,0.f,0.f};
#pragma unroll
    for (int i = 0; i < 2; ++i)
#pragma unroll
        for (int j = 0; j < 4; ++j) acc[i][j] = vzero;

    // Cooperative staging: thread t owns one 32B chunk of one row in each tile.
    const int la_r = tid >> 1;          // 0..127 (tile row)
    const int la_c = (tid & 1) * 16;    // 0 or 16 (k chunk)

    const bf16_t* gA  = A  + (size_t)(bm + la_r) * K + la_c;
    const bf16_t* gBT = BT + (size_t)(bn + la_r) * K + la_c;

    // LDS byte addresses (generic LDS pointer: offset lives in addr[31:0]).
    unsigned ldsA[2], ldsB[2];
#pragma unroll
    for (int b = 0; b < 2; ++b) {
        ldsA[b] = (unsigned)(size_t)&sA [b][la_r][la_c];
        ldsB[b] = (unsigned)(size_t)&sBT[b][la_r][la_c];
    }

    // Issue the 4 async-to-LDS b128 ops (per thread) staging K-tile `kb`.
    // INST_OFFSET is added to both the LDS and the global address.
    auto issue_tile = [&](int buf, int kb) {
        const bf16_t* pA = gA  + (size_t)kb * BK;
        const bf16_t* pB = gBT + (size_t)kb * BK;
        asm volatile("global_load_async_to_lds_b128 %0, %1, off"
                     :: "v"(ldsA[buf]), "v"(pA) : "memory");
        asm volatile("global_load_async_to_lds_b128 %0, %1, off offset:16"
                     :: "v"(ldsA[buf]), "v"(pA) : "memory");
        asm volatile("global_load_async_to_lds_b128 %0, %1, off"
                     :: "v"(ldsB[buf]), "v"(pB) : "memory");
        asm volatile("global_load_async_to_lds_b128 %0, %1, off offset:16"
                     :: "v"(ldsB[buf]), "v"(pB) : "memory");
    };

    const int nk = K / BK;
    issue_tile(0, 0);                    // prologue: 4 async ops in flight

    for (int kb = 0; kb < nk; ++kb) {
        const int buf = kb & 1;

        if (kb + 1 < nk) {
            issue_tile(buf ^ 1, kb + 1);                 // lookahead tile
            asm volatile("s_wait_asynccnt 0x4" ::: "memory"); // current tile done
        } else {
            asm volatile("s_wait_asynccnt 0x0" ::: "memory");
        }
        if (kb + 2 < nk)                                  // deep HBM prefetch
            __builtin_prefetch(gA + (size_t)(kb + 2) * BK, 0, 0);
        __syncthreads();

        // ---- fragments per documented 16-bit WMMA VGPR layouts ----
        // A (16x32): lane holds row m = lane%16; elems 0..7 = K ak..ak+7,
        //            elems 8..15 = K ak+16..ak+23 (ak = 0|8 by half-wave).
        FragU afr[2], bfr[4];
        const int ar = lane & 15;
        const int ak = (lane < 16) ? 0 : 8;
#pragma unroll
        for (int mt = 0; mt < 2; ++mt) {
            const bf16_t* p = &sA[buf][wm + mt * 16 + ar][0];
            afr[mt].q[0] = *(const uint4*)(p + ak);
            afr[mt].q[1] = *(const uint4*)(p + ak + 16);
        }
        // B (32x16): lane holds column n = lane%16; elems 0..7 = K bk0..bk0+7,
        //            elems 8..15 = K bk0+16..bk0+23 (bk0 = 0|8 by half-wave).
        const int bk0 = (lane < 16) ? 0 : 8;
#pragma unroll
        for (int nt = 0; nt < 4; ++nt) {
            const bf16_t* p = &sBT[buf][wn + nt * 16 + (lane & 15)][bk0];
            bfr[nt].q[0] = *(const uint4*)(p);
            bfr[nt].q[1] = *(const uint4*)(p + 16);
        }

#pragma unroll
        for (int mt = 0; mt < 2; ++mt)
#pragma unroll
            for (int nt = 0; nt < 4; ++nt)
                acc[mt][nt] = __builtin_amdgcn_wmma_f32_16x16x32_bf16(
                    /*neg_a=*/false, afr[mt].v,
                    /*neg_b=*/false, bfr[nt].v,
                    /*c_mod=*/(short)0, acc[mt][nt],
                    /*reuse_a=*/false, /*reuse_b=*/false);

        __syncthreads();   // buffer consumed -> safe to overwrite next round
    }

    // ---- store: v8f element r -> (m = r + 8*(lane>>4), n = lane&15) ----
    const int cn = lane & 15;
    const int cm = (lane >> 4) * 8;
#pragma unroll
    for (int mt = 0; mt < 2; ++mt)
#pragma unroll
        for (int nt = 0; nt < 4; ++nt) {
            float* pD = D + (size_t)(bm + wm + mt * 16 + cm) * N
                          + (bn + wn + nt * 16 + cn);
#pragma unroll
            for (int r = 0; r < 8; ++r)
                pD[(size_t)r * N] = acc[mt][nt][r];
        }
}

// ---------------------------------------------------------------------------
__global__ void k_f32_to_bf16(const float* __restrict__ in,
                              bf16_t* __restrict__ out, int n)
{
    int i = blockIdx.x * blockDim.x + threadIdx.x;
    if (i < n) out[i] = (bf16_t)in[i];
}

// out[c][r] = in[r][c]  (out is cols x rows, bf16)
__global__ void k_f32_to_bf16_T(const float* __restrict__ in,
                                bf16_t* __restrict__ out, int rows, int cols)
{
    int i = blockIdx.x * blockDim.x + threadIdx.x;
    if (i < rows * cols) {
        int r = i / cols, c = i % cols;
        out[(size_t)c * rows + r] = (bf16_t)in[i];
    }
}

// ---------------------------------------------------------------------------
extern "C" void kernel_launch(void* const* d_in, const int* in_sizes, int n_in,
                              void* d_out, int out_size, void* d_ws, size_t ws_size,
                              hipStream_t stream)
{
    const float* x = (const float*)d_in[0];   // (4,4096,1024)
    const float* A = (const float*)d_in[1];   // (1024,1024)
    const float* B = (const float*)d_in[2];
    const float* C = (const float*)d_in[3];
    float* y = (float*)d_out;                 // (4,4096,1024) f32

    const int Dm = 1024;
    const int M  = 4 * 4096;                  // 16384 fused (batch*seq) rows

    // workspace carve-out (~50 MB)
    char* ws = (char*)d_ws;
    size_t off = 0;
    bf16_t* Xb  = (bf16_t*)(ws + off); off += (size_t)M  * Dm * sizeof(bf16_t);
    bf16_t* Atb = (bf16_t*)(ws + off); off += (size_t)Dm * Dm * sizeof(bf16_t); // A^T
    bf16_t* Btb = (bf16_t*)(ws + off); off += (size_t)Dm * Dm * sizeof(bf16_t); // B^T
    bf16_t* Cb  = (bf16_t*)(ws + off); off += (size_t)Dm * Dm * sizeof(bf16_t); // C
    bf16_t* Tb  = (bf16_t*)(ws + off); off += (size_t)Dm * Dm * sizeof(bf16_t);
    bf16_t* Wtb = (bf16_t*)(ws + off); off += (size_t)Dm * Dm * sizeof(bf16_t); // W^T
    float*  Tf  = (float*) (ws + off); off += (size_t)Dm * Dm * sizeof(float);
    float*  Wf  = (float*) (ws + off); off += (size_t)Dm * Dm * sizeof(float);

    const int cvt = 256;

    // Downcasts; every GEMM B-operand is emitted pre-transposed (N x K).
    k_f32_to_bf16  <<<(M * Dm + cvt - 1) / cvt, cvt, 0, stream>>>(x, Xb, M * Dm);
    k_f32_to_bf16_T<<<(Dm * Dm + cvt - 1) / cvt, cvt, 0, stream>>>(A, Atb, Dm, Dm);
    k_f32_to_bf16_T<<<(Dm * Dm + cvt - 1) / cvt, cvt, 0, stream>>>(B, Btb, Dm, Dm);
    k_f32_to_bf16  <<<(Dm * Dm + cvt - 1) / cvt, cvt, 0, stream>>>(C, Cb, Dm * Dm);

    dim3 blk(THREADS);
    dim3 gW(Dm / BM, Dm / BN);   // 8 x 8

    // T = B^T @ A        (A-op: B^T [i,h];  BT-op: A^T [k,h])
    gemm_bf16_wmma<<<gW, blk, 0, stream>>>(Btb, Atb, Tf, Dm, Dm, Dm);
    k_f32_to_bf16 <<<(Dm * Dm + cvt - 1) / cvt, cvt, 0, stream>>>(Tf, Tb, Dm * Dm);

    // W = T @ C^T        (BT-op for C^T in NxK form is just C [o,k])
    gemm_bf16_wmma<<<gW, blk, 0, stream>>>(Tb, Cb, Wf, Dm, Dm, Dm);
    k_f32_to_bf16_T<<<(Dm * Dm + cvt - 1) / cvt, cvt, 0, stream>>>(Wf, Wtb, Dm, Dm); // W^T

    // y = x @ W   (16384 x 1024 x 1024; x streams from HBM, W^T L2-resident)
    dim3 gY(M / BM, Dm / BN);    // 128 x 8
    gemm_bf16_wmma<<<gY, blk, 0, stream>>>(Xb, Wtb, y, M, Dm, Dm);
}
